// network_66391604462173
// MI455X (gfx1250) — compile-verified, matrix operand order checked by reading, not run on hardware
//
#include <hip/hip_runtime.h>

// ---------------------------------------------------------------------------
// CDNA5 (gfx1250) implementation of the 3-level PWC-style 3D flow network.
// All 3x3x3 convs and 4x4x4 stride-2 transposed convs run as implicit GEMM on
// v_wmma_f32_16x16x32_f16 (f16 A/B, fp32 accum). K is padded per input
// channel so the tap index of every WMMA fragment element is loop-invariant.
// All gathers use always-valid clamped offsets + 0/1 float multipliers so the
// inner loop is pure unconditional loads + VALU + WMMA (no exec-mask branching).
// ---------------------------------------------------------------------------

typedef __attribute__((ext_vector_type(16))) _Float16 v16h;
typedef __attribute__((ext_vector_type(8)))  float    v8f;

// ---------------------------------------------------------------------------
// Implicit-GEMM conv3d, k=3, pad=1, stride in {1,2}. NCDHW, B=1.
// One wave per 16(Mvox) x 16(Cout) tile. Logical K = Cin*32: k = ci*32 + tap,
// taps 27..31 are zero padding; tap(i,hi) is constant across the K loop.
// A fragment (16-bit 16x32): lanes 0-15 hold K {0..7,16..23}; lanes 16-31 hold
// K {8..15,24..31}. B mirrors with N = lane&15. D: vgpr r holds
// M = r + 8*(lane>=16), N = lane&15.
// ---------------------------------------------------------------------------
__global__ void conv3d_wmma_kernel(const float* __restrict__ in,
                                   const float* __restrict__ wgt,
                                   const float* __restrict__ bias,
                                   float* __restrict__ out,
                                   int Cin, int Cout, int Si, int So, int stride) {
  const int lane = threadIdx.x;
  const int row  = lane & 15;
  const int hi   = lane >> 4;
  const long long Si3 = (long long)Si * Si * Si;
  const long long So3 = (long long)So * So * So;

  const long long m = (long long)blockIdx.x * 16 + row;   // output voxel (A row)
  const int       n = blockIdx.y * 16 + row;              // output channel (B col)
  const bool mvalid = m < So3;
  int ow = 0, oh = 0, od = 0;
  if (mvalid) {
    ow = (int)(m % So);
    oh = (int)((m / So) % So);
    od = (int)(m / ((long long)So * So));
  }
  const bool nvalid = n < Cout;

  // ---- loop-invariant per-element precompute ------------------------------
  int   aoff[16];   // spatial offset within a channel slab (always in-bounds)
  float amul[16];   // 1.0 if element contributes, else 0.0
  int   woff[16];   // n*Cin*27 + tap (always in-bounds)
  float wmul[16];
#pragma unroll
  for (int i = 0; i < 16; ++i) {
    const int tap = hi * 8 + ((i < 8) ? i : (8 + i));   // 0..31, constant over K
    const bool tv = tap < 27;
    const int kd = tap / 9, kh = (tap / 3) % 3, kw = tap % 3;
    const int id = od * stride + kd - 1;
    const int ih = oh * stride + kh - 1;
    const int iw = ow * stride + kw - 1;
    const bool av = tv && mvalid &&
                    id >= 0 && id < Si && ih >= 0 && ih < Si && iw >= 0 && iw < Si;
    amul[i] = av ? 1.f : 0.f;
    aoff[i] = av ? (((id * Si) + ih) * Si + iw) : 0;
    const bool wv = tv && nvalid;
    wmul[i] = wv ? 1.f : 0.f;
    woff[i] = wv ? (n * Cin * 27 + tap) : 0;
  }

  // ---- K loop: one 32-chunk per input channel -----------------------------
  v8f acc = {};
  const float* __restrict__ slab = in;    // advances by Si3 per channel
  const float* __restrict__ wcur = wgt;   // advances by 27 per channel
  for (int ci = 0; ci < Cin; ++ci) {
    if (ci + 1 < Cin) __builtin_prefetch(slab + Si3, 0, 0);
    v16h a, b;
#pragma unroll
    for (int i = 0; i < 16; ++i) {
      const float ra = slab[aoff[i]];          // unconditional, address valid
      const float rb = wcur[woff[i]];
      a[i] = (_Float16)(ra * amul[i]);
      b[i] = (_Float16)(rb * wmul[i]);
    }
    acc = __builtin_amdgcn_wmma_f32_16x16x32_f16(false, a, false, b,
                                                 (short)0, acc, false, false);
    slab += Si3;
    wcur += 27;
  }

  const float bn = nvalid ? bias[n] : 0.f;
#pragma unroll
  for (int r = 0; r < 8; ++r) {
    const long long mm = (long long)blockIdx.x * 16 + r + hi * 8;
    if (mm < So3 && nvalid)
      out[(long long)n * So3 + mm] = acc[r] + bn;
  }
}

// ---------------------------------------------------------------------------
// Implicit-GEMM ConvTranspose3d, k=4, s=2, p=1 (So = 2*Si). Outputs split into
// 8 parity classes (grid.z) so kernel taps are tile-uniform: for parity p,
// o = 2*j + 1 - p, kd = p + 2*t, i = j + 1 - p - t. K = Cin*8: tap = i&7 is
// loop-invariant, ci = 4*cb + hi + (i>=8 ? 2 : 0). Weights (Cin,Cout,4,4,4).
// Full 4-channel chunks run check-free; one optional epilogue chunk clamps.
// ---------------------------------------------------------------------------
__global__ void deconv3d_wmma_kernel(const float* __restrict__ in,
                                     const float* __restrict__ wgt,
                                     const float* __restrict__ bias,
                                     float* __restrict__ out,
                                     int Cin, int Cout, int Si) {
  const int So = Si * 2;
  const int lane = threadIdx.x;
  const int row  = lane & 15;
  const int hi   = lane >> 4;
  const int Si3  = Si * Si * Si;              // <= 262144, fits int
  const long long So3 = (long long)So * So * So;
  const int pd = (blockIdx.z >> 2) & 1, ph = (blockIdx.z >> 1) & 1, pw = blockIdx.z & 1;

  const int m = blockIdx.x * 16 + row;        // subgrid voxel index
  const bool mvalid = m < Si3;
  int jw = 0, jh = 0, jd = 0;
  if (mvalid) {
    jw = m % Si;
    jh = (m / Si) % Si;
    jd = m / (Si * Si);
  }
  const int n = blockIdx.y * 16 + row;
  const bool nvalid = n < Cout;
  const int wstride = Cout * 64;              // weight elems per input channel

  // ---- loop-invariant per-element precompute ------------------------------
  int   poff[16];   // cio*Si3 + spatial offset (valid within 4-ch chunk)
  float amul[16];
  int   wpoff[16];  // cio*wstride + n*64 + kd*16 + kh*4 + kw
  float wmul[16];
  int   cio[16];    // channel offset within a 4-channel chunk (0..3)
#pragma unroll
  for (int i = 0; i < 16; ++i) {
    const int tap = i & 7;                       // constant over K loop
    const int co  = hi + ((i < 8) ? 0 : 2);
    cio[i] = co;
    const int td = (tap >> 2) & 1, th = (tap >> 1) & 1, tw = tap & 1;
    const int id = jd + 1 - pd - td;
    const int ih = jh + 1 - ph - th;
    const int iw = jw + 1 - pw - tw;
    const bool av = mvalid &&
                    id >= 0 && id < Si && ih >= 0 && ih < Si && iw >= 0 && iw < Si;
    amul[i] = av ? 1.f : 0.f;
    const int sp = av ? (((id * Si) + ih) * Si + iw) : 0;
    poff[i] = co * Si3 + sp;
    const int kd = pd + 2 * td, kh = ph + 2 * th, kw = pw + 2 * tw;
    wmul[i]  = nvalid ? 1.f : 0.f;
    wpoff[i] = co * wstride + (nvalid ? (n * 64 + kd * 16 + kh * 4 + kw) : 0);
  }

  // ---- K loop: 4 input channels per 32-chunk ------------------------------
  v8f acc = {};
  const float* __restrict__ basep = in;
  const float* __restrict__ wbase = wgt;
  const int nfull = Cin >> 2;
  for (int cb = 0; cb < nfull; ++cb) {
    v16h a, b;
#pragma unroll
    for (int i = 0; i < 16; ++i) {
      const float ra = basep[poff[i]];
      const float rb = wbase[wpoff[i]];
      a[i] = (_Float16)(ra * amul[i]);
      b[i] = (_Float16)(rb * wmul[i]);
    }
    acc = __builtin_amdgcn_wmma_f32_16x16x32_f16(false, a, false, b,
                                                 (short)0, acc, false, false);
    basep += 4 * Si3;
    wbase += 4 * wstride;
  }
  if (Cin & 3) {                               // epilogue chunk with clamping
    const int cbase = nfull * 4;
    v16h a, b;
#pragma unroll
    for (int i = 0; i < 16; ++i) {
      const bool cv = (cbase + cio[i]) < Cin;
      const float cm = cv ? 1.f : 0.f;
      const int  pa = cv ? poff[i] : 0;
      const int  pb = cv ? wpoff[i] : 0;
      const float ra = basep[pa];
      const float rb = wbase[pb];
      a[i] = (_Float16)(ra * (amul[i] * cm));
      b[i] = (_Float16)(rb * (wmul[i] * cm));
    }
    acc = __builtin_amdgcn_wmma_f32_16x16x32_f16(false, a, false, b,
                                                 (short)0, acc, false, false);
  }

  const float bn = nvalid ? bias[n] : 0.f;
#pragma unroll
  for (int r = 0; r < 8; ++r) {
    const int mm = blockIdx.x * 16 + r + hi * 8;
    if (mm < Si3 && nvalid) {
      const int jw2 = mm % Si;
      const int jh2 = (mm / Si) % Si;
      const int jd2 = mm / (Si * Si);
      const int od = 2 * jd2 + 1 - pd;
      const int oh = 2 * jh2 + 1 - ph;
      const int ow = 2 * jw2 + 1 - pw;
      out[(long long)n * So3 + ((long long)od * So + oh) * So + ow] = acc[r] + bn;
    }
  }
}

// ---------------------------------------------------------------------------
// InstanceNorm3d (affine=False) + LeakyReLU(0.1), in place. One block per ch.
// ---------------------------------------------------------------------------
__global__ void inorm_leaky_kernel(float* __restrict__ buf, int Nsp) {
  const int c = blockIdx.x;
  float* p = buf + (size_t)c * Nsp;
  __shared__ float s1[256];
  __shared__ float s2[256];
  float a = 0.f, b = 0.f;
  for (int i = threadIdx.x; i < Nsp; i += blockDim.x) {
    const float x = p[i];
    a += x;
    b += x * x;
  }
  s1[threadIdx.x] = a;
  s2[threadIdx.x] = b;
  __syncthreads();
  for (int s = 128; s > 0; s >>= 1) {
    if ((int)threadIdx.x < s) {
      s1[threadIdx.x] += s1[threadIdx.x + s];
      s2[threadIdx.x] += s2[threadIdx.x + s];
    }
    __syncthreads();
  }
  const float mean = s1[0] / (float)Nsp;
  const float var  = s2[0] / (float)Nsp - mean * mean;
  const float inv  = rsqrtf(var + 1e-5f);
  for (int i = threadIdx.x; i < Nsp; i += blockDim.x) {
    const float y = (p[i] - mean) * inv;
    p[i] = (y >= 0.f) ? y : 0.1f * y;
  }
}

// ---------------------------------------------------------------------------
// Cost volume (r=2 -> 125 shifts) with fused LeakyReLU(0.1). Zero-padded OOB.
// Shift order: v (D) slowest, then h (H), then d (W)  -> matches reference.
// ---------------------------------------------------------------------------
__global__ void costvol_leaky_kernel(const float* __restrict__ f1,
                                     const float* __restrict__ f2,
                                     float* __restrict__ out, int C, int S) {
  const long long S3 = (long long)S * S * S;
  const long long idx = (long long)blockIdx.x * blockDim.x + threadIdx.x;
  if (idx >= 125 * S3) return;
  const int sh = (int)(idx / S3);
  const long long p = idx % S3;
  const int pw = (int)(p % S);
  const int ph = (int)((p / S) % S);
  const int pd = (int)(p / ((long long)S * S));
  const int dv = sh / 25 - 2;
  const int dh = (sh / 5) % 5 - 2;
  const int dw = sh % 5 - 2;
  const int qd = pd + dv, qh = ph + dh, qw = pw + dw;
  float acc = 0.f;
  if (qd >= 0 && qd < S && qh >= 0 && qh < S && qw >= 0 && qw < S) {
    const long long q = ((long long)qd * S + qh) * S + qw;
    for (int c = 0; c < C; ++c)
      acc += f1[c * S3 + p] * f2[c * S3 + q];
  }
  acc /= (float)C;
  out[(long long)sh * S3 + p] = (acc >= 0.f) ? acc : 0.1f * acc;
}

// ---------------------------------------------------------------------------
// Dense 3D warp: trilinear sample at (grid + flow), coords clamped to border.
// flow layout (3, S^3): ch0 = d-offset, ch1 = h, ch2 = w.
// ---------------------------------------------------------------------------
__global__ void warp3d_kernel(const float* __restrict__ img,
                              const float* __restrict__ flow,
                              float* __restrict__ out, int C, int S) {
  const long long S3 = (long long)S * S * S;
  const long long idx = (long long)blockIdx.x * blockDim.x + threadIdx.x;
  if (idx >= (long long)C * S3) return;
  const int c = (int)(idx / S3);
  const long long p = idx % S3;
  const int w = (int)(p % S);
  const int h = (int)((p / S) % S);
  const int d = (int)(p / ((long long)S * S));

  const float lim = (float)(S - 1);
  const float cd = fminf(fmaxf((float)d + flow[p], 0.f), lim);
  const float ch = fminf(fmaxf((float)h + flow[S3 + p], 0.f), lim);
  const float cw = fminf(fmaxf((float)w + flow[2 * S3 + p], 0.f), lim);

  const int d0 = (int)cd, h0 = (int)ch, w0 = (int)cw;
  const float fd = cd - (float)d0, fh = ch - (float)h0, fw = cw - (float)w0;
  const int d1 = (d0 + 1 > S - 1) ? S - 1 : d0 + 1;
  const int h1 = (h0 + 1 > S - 1) ? S - 1 : h0 + 1;
  const int w1 = (w0 + 1 > S - 1) ? S - 1 : w0 + 1;

  const float* im = img + (long long)c * S3;
#define AT(dd, hh, ww) im[((long long)(dd) * S + (hh)) * S + (ww)]
  const float c00 = AT(d0, h0, w0) * (1.f - fw) + AT(d0, h0, w1) * fw;
  const float c01 = AT(d0, h1, w0) * (1.f - fw) + AT(d0, h1, w1) * fw;
  const float c10 = AT(d1, h0, w0) * (1.f - fw) + AT(d1, h0, w1) * fw;
  const float c11 = AT(d1, h1, w0) * (1.f - fw) + AT(d1, h1, w1) * fw;
#undef AT
  const float c0 = c00 * (1.f - fh) + c01 * fh;
  const float c1 = c10 * (1.f - fh) + c11 * fh;
  out[(long long)c * S3 + p] = c0 * (1.f - fd) + c1 * fd;
}

// ---------------------------------------------------------------------------
static inline unsigned cdivu(long long a, long long b) { return (unsigned)((a + b - 1) / b); }

extern "C" void kernel_launch(void* const* d_in, const int* in_sizes, int n_in,
                              void* d_out, int out_size, void* d_ws, size_t ws_size,
                              hipStream_t stream) {
  (void)in_sizes; (void)n_in; (void)out_size;

  const float* atlas  = (const float*)d_in[0];
  const float* target = (const float*)d_in[1];
  const float* W_ct = (const float*)d_in[2];   const float* b_ct = (const float*)d_in[3];
  const float* W1a  = (const float*)d_in[4];   const float* b1a  = (const float*)d_in[5];
  const float* W1aa = (const float*)d_in[6];   const float* b1aa = (const float*)d_in[7];
  const float* W1b  = (const float*)d_in[8];   const float* b1b  = (const float*)d_in[9];
  const float* W2a  = (const float*)d_in[10];  const float* b2a  = (const float*)d_in[11];
  const float* W2aa = (const float*)d_in[12];  const float* b2aa = (const float*)d_in[13];
  const float* W2b  = (const float*)d_in[14];  const float* b2b  = (const float*)d_in[15];
  const float* W20  = (const float*)d_in[16];  const float* b20  = (const float*)d_in[17];
  const float* Wpf2 = (const float*)d_in[18];  const float* bpf2 = (const float*)d_in[19];
  const float* Wd2  = (const float*)d_in[20];  const float* bd2  = (const float*)d_in[21];
  const float* Wuf2 = (const float*)d_in[22];  const float* buf2 = (const float*)d_in[23];
  const float* W10  = (const float*)d_in[24];  const float* b10  = (const float*)d_in[25];
  const float* Wpf1 = (const float*)d_in[26];  const float* bpf1 = (const float*)d_in[27];
  const float* Wd1  = (const float*)d_in[28];  const float* bd1  = (const float*)d_in[29];
  const float* Wuf1 = (const float*)d_in[30];  const float* buf1 = (const float*)d_in[31];
  const float* W00  = (const float*)d_in[32];  const float* b00  = (const float*)d_in[33];
  const float* Wpf0 = (const float*)d_in[34];  const float* bpf0 = (const float*)d_in[35];
  const float* Wd0  = (const float*)d_in[36];  const float* bd0  = (const float*)d_in[37];

  const long long V128 = 128LL * 128 * 128;
  const long long V64  = 64LL * 64 * 64;
  const long long V32  = 32LL * 32 * 32;
  const long long V16  = 16LL * 16 * 16;

  // Bump-allocate workspace (NCDHW fp32, B=1). Concats are contiguous channel
  // stacks, so producers write directly at channel offsets.
  float* ws = (float*)d_ws;
  size_t off = 0;
  auto alloc = [&](long long elems) { float* p = ws + off; off += (size_t)elems; return p; };
  float* im1  = alloc(4 * V64);
  float* im2  = alloc(4 * V64);
  float* tA   = alloc(16 * V32);
  float* tB   = alloc(16 * V32);
  float* c11  = alloc(16 * V32);
  float* c21  = alloc(16 * V32);
  float* sA   = alloc(32 * V16);
  float* sB   = alloc(32 * V16);
  float* c12  = alloc(32 * V16);
  float* c22  = alloc(32 * V16);
  float* cv2  = alloc(125 * V16);
  float* x2   = alloc(32 * V16);
  float* fl2  = alloc(3 * V16);
  float* cat1 = alloc(147 * V32);   // [corr1 | c11 | up_flow2 | up_feat2]
  float* x1   = alloc(32 * V32);
  float* fl1  = alloc(3 * V32);
  float* wp1  = alloc(16 * V32);
  float* cat0 = alloc(196 * V64);   // [corr0 | im1 | up_flow1 | up_feat1]
  float* wp0  = alloc(4 * V64);
  float* x0   = alloc(32 * V64);
  float* fl0  = alloc(3 * V64);
  if (off * sizeof(float) > ws_size) return;   // scratch too small: do nothing

  auto conv = [&](const float* in, const float* w, const float* b, float* out,
                  int Cin, int Cout, int Si, int So, int stride) {
    dim3 g(cdivu((long long)So * So * So, 16), cdivu(Cout, 16), 1);
    conv3d_wmma_kernel<<<g, dim3(32), 0, stream>>>(in, w, b, out, Cin, Cout, Si, So, stride);
  };
  auto deconv = [&](const float* in, const float* w, const float* b, float* out,
                    int Cin, int Cout, int Si) {
    dim3 g(cdivu((long long)Si * Si * Si, 16), cdivu(Cout, 16), 8);
    deconv3d_wmma_kernel<<<g, dim3(32), 0, stream>>>(in, w, b, out, Cin, Cout, Si);
  };
  auto inorm = [&](float* buf, int C, long long Nsp) {
    inorm_leaky_kernel<<<dim3(C), dim3(256), 0, stream>>>(buf, (int)Nsp);
  };
  auto costvol = [&](const float* f1, const float* f2, float* out, int C, int S) {
    const long long total = 125LL * S * S * S;
    costvol_leaky_kernel<<<dim3(cdivu(total, 256)), dim3(256), 0, stream>>>(f1, f2, out, C, S);
  };
  auto warp = [&](const float* img, const float* flow, float* out, int C, int S) {
    const long long total = (long long)C * S * S * S;
    warp3d_kernel<<<dim3(cdivu(total, 256)), dim3(256), 0, stream>>>(img, flow, out, C, S);
  };

  // ---- shared CT encoder --------------------------------------------------
  conv(atlas,  W_ct, b_ct, im2, 1, 4, 128, 64, 2);  inorm(im2, 4, V64);
  conv(target, W_ct, b_ct, im1, 1, 4, 128, 64, 2);  inorm(im1, 4, V64);

  conv(im1, W1a,  b1a,  tA,  4, 16, 64, 32, 2);  inorm(tA, 16, V32);
  conv(tA,  W1aa, b1aa, tB, 16, 16, 32, 32, 1);  inorm(tB, 16, V32);
  conv(tB,  W1b,  b1b,  c11,16, 16, 32, 32, 1);  inorm(c11, 16, V32);

  conv(im2, W1a,  b1a,  tA,  4, 16, 64, 32, 2);  inorm(tA, 16, V32);
  conv(tA,  W1aa, b1aa, tB, 16, 16, 32, 32, 1);  inorm(tB, 16, V32);
  conv(tB,  W1b,  b1b,  c21,16, 16, 32, 32, 1);  inorm(c21, 16, V32);

  conv(c11, W2a,  b2a,  sA, 16, 32, 32, 16, 2);  inorm(sA, 32, V16);
  conv(sA,  W2aa, b2aa, sB, 32, 32, 16, 16, 1);  inorm(sB, 32, V16);
  conv(sB,  W2b,  b2b,  c12,32, 32, 16, 16, 1);  inorm(c12, 32, V16);

  conv(c21, W2a,  b2a,  sA, 16, 32, 32, 16, 2);  inorm(sA, 32, V16);
  conv(sA,  W2aa, b2aa, sB, 32, 32, 16, 16, 1);  inorm(sB, 32, V16);
  conv(sB,  W2b,  b2b,  c22,32, 32, 16, 16, 1);  inorm(c22, 32, V16);

  // ---- level 2 ------------------------------------------------------------
  costvol(c12, c22, cv2, 32, 16);
  conv(cv2, W20, b20, x2, 125, 32, 16, 16, 1);  inorm(x2, 32, V16);
  conv(x2, Wpf2, bpf2, fl2, 32, 3, 16, 16, 1);
  float* upfl2 = cat1 + 141 * V32;
  float* upft2 = cat1 + 144 * V32;
  deconv(fl2, Wd2, bd2, upfl2, 3, 3, 16);      // 3 x 32^3
  deconv(x2, Wuf2, buf2, upft2, 32, 3, 16);    // 3 x 32^3

  // ---- level 1 ------------------------------------------------------------
  warp(c21, upfl2, wp1, 16, 32);
  costvol(c11, wp1, cat1, 16, 32);             // corr1 at channel 0
  hipMemcpyAsync(cat1 + 125 * V32, c11, sizeof(float) * 16 * V32,
                 hipMemcpyDeviceToDevice, stream);
  conv(cat1, W10, b10, x1, 147, 32, 32, 32, 1);  inorm(x1, 32, V32);
  conv(x1, Wpf1, bpf1, fl1, 32, 3, 32, 32, 1);
  float* upfl1 = cat0 + 129 * V64;
  float* upft1 = cat0 + 132 * V64;
  deconv(fl1, Wd1, bd1, upfl1, 3, 3, 32);      // 3 x 64^3
  deconv(x1, Wuf1, buf1, upft1, 32, 64, 32);   // 64 x 64^3

  // ---- level 0 ------------------------------------------------------------
  warp(im2, upfl1, wp0, 4, 64);
  costvol(im1, wp0, cat0, 4, 64);              // corr0 at channel 0
  hipMemcpyAsync(cat0 + 125 * V64, im1, sizeof(float) * 4 * V64,
                 hipMemcpyDeviceToDevice, stream);
  conv(cat0, W00, b00, x0, 196, 32, 64, 64, 1);  inorm(x0, 32, V64);
  conv(x0, Wpf0, bpf0, fl0, 32, 3, 64, 64, 1);

  // ---- outputs: (flow, c22, c12) -----------------------------------------
  float* outp = (float*)d_out;
  deconv(fl0, Wd0, bd0, outp, 3, 3, 64);       // flow: 3 x 128^3
  hipMemcpyAsync(outp + 3 * V128, c22, sizeof(float) * 32 * V16,
                 hipMemcpyDeviceToDevice, stream);
  hipMemcpyAsync(outp + 3 * V128 + 32 * V16, c12, sizeof(float) * 32 * V16,
                 hipMemcpyDeviceToDevice, stream);
}
